// WindowAttention_80865644249598
// MI455X (gfx1250) — compile-verified, matrix operand order checked by reading, not run on hardware
//
#include <hip/hip_runtime.h>
#include <hip/hip_bf16.h>

typedef __attribute__((ext_vector_type(16))) __bf16 v16bf;
typedef __attribute__((ext_vector_type(8)))  float  v8f;
typedef __attribute__((ext_vector_type(4)))  unsigned int u32x4;
typedef __attribute__((ext_vector_type(8)))  unsigned int u32x8;

#define kB  1024
#define kN  128
#define kC  128
#define kH  8
#define kHD 16
#define kNW 64
#define kScale 0.25f   // hd^-0.5, folded into packed Wq

// K-index pattern for 16-bit A/B operands of v_wmma_*_16x16x32 (ISA 7.12.2):
// lane half sel = lane/16, vector element e in [0,16)
__device__ __forceinline__ int klocal(int sel, int e) {
  return e + sel * 8 + (e >= 8 ? 8 : 0);
}

// ---------------------------------------------------------------------------
// TDM: issue a 2-D tile load (global -> LDS) through the Tensor Data Mover.
// Builds D# group0 (4 SGPRs) + group1 (8 SGPRs) per CDNA5 ISA
// and issues the 2-operand (<=2-D tensor) form of tensor_load_to_lds.
// Tracked by TENSORcnt. EXEC is ignored; call from ONE wave, then
// s_wait_tensorcnt + workgroup barrier before any wave reads the LDS tile.
//   data_size_code: 1 = 2-byte elems, 2 = 4-byte elems
//   width  : elements per row (tile_dim0 == tensor_dim0)
//   rows   : number of rows   (tile_dim1 == tensor_dim1)
//   stride : row stride in elements (tensor_dim0_stride)
// ---------------------------------------------------------------------------
__device__ __forceinline__ void tdm_load_2d(unsigned lds_addr, const void* gptr,
                                            unsigned data_size_code,
                                            unsigned width, unsigned rows,
                                            unsigned stride) {
  unsigned long long ga = (unsigned long long)(uintptr_t)gptr;
  u32x4 g0;
  g0[0] = 1u;                                   // count=1, user descriptor
  g0[1] = lds_addr;                             // LDS byte address
  g0[2] = (unsigned)ga;                         // global_addr[31:0]
  g0[3] = (unsigned)((ga >> 32) & 0x01FFFFFFu)  // global_addr[56:32]
          | 0x80000000u;                        // type=2 ("image")
  u32x8 g1;
  g1[0] = data_size_code << 16;                 // wg_mask=0, data_size
  g1[1] = (width & 0xFFFFu) << 16;              // tensor_dim0[15:0]
  g1[2] = (width >> 16) | ((rows & 0xFFFFu) << 16);   // dim0 hi, dim1 lo
  g1[3] = (rows >> 16) | ((width & 0xFFFFu) << 16);   // dim1 hi, tile_dim0
  g1[4] = rows & 0xFFFFu;                       // tile_dim1, tile_dim2=0
  g1[5] = stride;                               // tensor_dim0_stride[31:0]
  g1[6] = 0u;                                   // stride hi, dim1_stride lo
  g1[7] = 0u;
  asm volatile("tensor_load_to_lds %0, %1" : : "s"(g0), "s"(g1) : "memory");
}

// ---------------------------------------------------------------------------
// Kernel 0: pack weights into WMMA B-operand layout (bf16) + gather rel bias
// Wpack:     [24 ntile][4 kc][32 lane][16 e]  for [Wq(scaled) | Wkv]
// Wprojpack: [ 8 ntile][4 kc][32 lane][16 e]
// bias_full: [h][r][c] = bias_table[rel_index[r][c]][h]
// ---------------------------------------------------------------------------
__global__ void prep_kernel(const float* __restrict__ Wq,
                            const float* __restrict__ Wkv,
                            const float* __restrict__ Wproj,
                            const float* __restrict__ bias_table,
                            const int*   __restrict__ rel_index,
                            __bf16* __restrict__ Wpack,
                            __bf16* __restrict__ Wprojpack,
                            float*  __restrict__ bias_full) {
  int tid = blockIdx.x * blockDim.x + threadIdx.x;
  const int NPACK = 24 * 4 * 32;
  const int NPPK  = 8 * 4 * 32;
  if (tid < NPACK) {
    int lane = tid & 31, kc = (tid >> 5) & 3, t = tid >> 7;
    int sel = lane >> 4, ncol = t * 16 + (lane & 15);
    __bf16* dst = Wpack + (size_t)tid * 16;
    for (int e = 0; e < 16; ++e) {
      int k = kc * 32 + klocal(sel, e);
      float w = (ncol < 128) ? (Wq[k * 128 + ncol] * kScale)
                             : Wkv[k * 256 + (ncol - 128)];
      dst[e] = (__bf16)w;
    }
  } else if (tid < NPACK + NPPK) {
    int id = tid - NPACK;
    int lane = id & 31, kc = (id >> 5) & 3, t = id >> 7;
    int sel = lane >> 4, ncol = t * 16 + (lane & 15);
    __bf16* dst = Wprojpack + (size_t)id * 16;
    for (int e = 0; e < 16; ++e) {
      int k = kc * 32 + klocal(sel, e);
      dst[e] = (__bf16)Wproj[k * 128 + ncol];
    }
  } else {
    int id = tid - (NPACK + NPPK);
    if (id < kH * kN * kN) {
      int h = id / (kN * kN);
      int rc = id % (kN * kN);
      bias_full[id] = bias_table[rel_index[rc] * kH + h];
    }
  }
}

// ---------------------------------------------------------------------------
// Kernel 1: fused QKV projection.  (B*N x C) @ [Wq|Wkv] (C x 384)
// Per block: 8 waves x 16-row A tiles; 24 N-tiles; K=128 = 4x bf16 WMMA.
// All 4 B operands loaded as one clause, then 4 back-to-back WMMAs;
// next tile's weights prefetched (global_prefetch_b8).
// q (pre-scaled) and k stored bf16 (B,N,C layout), v stored f32.
// ---------------------------------------------------------------------------
__global__ void qkv_kernel(const float* __restrict__ x,
                           const __bf16* __restrict__ Wpack,
                           __bf16* __restrict__ qbf,
                           __bf16* __restrict__ kbf,
                           float*  __restrict__ vf) {
  __shared__ float xs[8][16 * 128];
  int tid = threadIdx.x, w = tid >> 5, lane = tid & 31;
  int rowbase = blockIdx.x * 128 + w * 16;

  // stage this wave's 16x128 f32 tile (8KB, contiguous) into LDS
  const float4* src = (const float4*)(x + (size_t)rowbase * kC);
  float4* dst4 = (float4*)xs[w];
  for (int i = lane; i < 16 * 128 / 4; i += 32) dst4[i] = src[i];
  __syncthreads();

  int sel = lane >> 4, r = lane & 15;
  v16bf A[4];
  for (int kc = 0; kc < 4; ++kc)
    for (int e = 0; e < 16; ++e)
      A[kc][e] = (__bf16)xs[w][r * 128 + kc * 32 + klocal(sel, e)];

  for (int t = 0; t < 24; ++t) {
    int tn = (t + 1 < 24) ? t + 1 : t;
    __builtin_prefetch(Wpack + (((size_t)tn * 4) * 32 + lane) * 16, 0, 0);
    v16bf Bv[4];
#pragma unroll
    for (int kc = 0; kc < 4; ++kc)
      Bv[kc] = *(const v16bf*)(Wpack + (((size_t)t * 4 + kc) * 32 + lane) * 16);
    v8f acc = {};
#pragma unroll
    for (int kc = 0; kc < 4; ++kc)
      acc = __builtin_amdgcn_wmma_f32_16x16x32_bf16(false, A[kc], false, Bv[kc],
                                                    (short)0, acc, false, false);
    int coln = t * 16 + r;
#pragma unroll
    for (int i = 0; i < 8; ++i) {
      int rg = rowbase + i + sel * 8;
      float res = acc[i];
      if (coln < 128)      qbf[(size_t)rg * kC + coln]         = (__bf16)res;
      else if (coln < 256) kbf[(size_t)rg * kC + (coln - 128)] = (__bf16)res;
      else                 vf [(size_t)rg * kC + (coln - 256)] = res;
    }
  }
}

// ---------------------------------------------------------------------------
// Kernel 2: fused attention per (b, head):
//   q/k/v head tiles staged to LDS by the Tensor Data Mover (wave 0 issues
//   3 D# loads, s_wait_tensorcnt, workgroup barrier), then
//   S = q k^T (WMMA, K=16 zero-padded to 32) + bias + mask -> softmax ->
//   top-2 (values+indices) -> out = sv0*v[si0] + sv1*v[si1]  (bf16 xo)
// Attention matrix lives entirely in accumulator VGPRs (never spilled).
// ---------------------------------------------------------------------------
__global__ void attn_kernel(const float* __restrict__ mask,
                            const float* __restrict__ bias_full,
                            const __bf16* __restrict__ qbf,
                            const __bf16* __restrict__ kbf,
                            const float* __restrict__ vf,
                            __bf16* __restrict__ xo) {
  __shared__ __bf16 qs[128][16];
  __shared__ __bf16 ks[128][16];
  __shared__ float  vs[128][16];
  int head = blockIdx.x, b = blockIdx.y;
  int tid = threadIdx.x, w = tid >> 5, lane = tid & 31;

  // TDM staging: 2-D tiles, row stride C elements, width = one head slice
  if (w == 0) {
    size_t rowoff = (size_t)b * kN * kC + head * kHD;
    tdm_load_2d((unsigned)(uintptr_t)&qs[0][0], qbf + rowoff, 1u, 16u, 128u, 128u);
    tdm_load_2d((unsigned)(uintptr_t)&ks[0][0], kbf + rowoff, 1u, 16u, 128u, 128u);
    tdm_load_2d((unsigned)(uintptr_t)&vs[0][0], vf  + rowoff, 2u, 16u, 128u, 128u);
    __builtin_amdgcn_s_wait_tensorcnt(0);
  }
  __syncthreads();

  int sel = lane >> 4, r = lane & 15;
  // A = q rows (K = hd = 16, upper half of K=32 zero-padded)
  v16bf Aq;
  for (int e = 0; e < 16; ++e) {
    int k = klocal(sel, e);
    Aq[e] = (k < 16) ? qs[w * 16 + r][k] : (__bf16)0.0f;
  }
  v8f acc[8];
  for (int j = 0; j < 8; ++j) {
    v16bf Bk;  // B = k^T tile: B[kk][n] = ks[n][kk]
    for (int e = 0; e < 16; ++e) {
      int k = klocal(sel, e);
      Bk[e] = (k < 16) ? ks[j * 16 + r][k] : (__bf16)0.0f;
    }
    v8f z = {};
    acc[j] = __builtin_amdgcn_wmma_f32_16x16x32_bf16(false, Aq, false, Bk,
                                                     (short)0, z, false, false);
  }

  const float* bh = bias_full + (size_t)head * kN * kN;
  const float* mw = mask + (size_t)(b % kNW) * kN * kN;
  for (int i = 0; i < 8; ++i) {
    int rg = w * 16 + i + sel * 8;
    for (int j = 0; j < 8; ++j) {
      int col = j * 16 + r;
      acc[j][i] += bh[rg * kN + col] + mw[rg * kN + col];
    }
  }

  // per-row softmax + top-2 + weighted V gather (row split over 16 lanes)
  for (int i = 0; i < 8; ++i) {
    int rg = w * 16 + i + sel * 8;
    float m = -3.0e38f;
    for (int j = 0; j < 8; ++j) m = fmaxf(m, acc[j][i]);
    for (int off = 1; off < 16; off <<= 1) m = fmaxf(m, __shfl_xor(m, off, 32));
    float pv[8], s = 0.f;
    for (int j = 0; j < 8; ++j) { pv[j] = __expf(acc[j][i] - m); s += pv[j]; }
    for (int off = 1; off < 16; off <<= 1) s += __shfl_xor(s, off, 32);
    float inv = 1.0f / s;

    float v1 = -1.f, v2 = -1.f; int i1 = 0, i2 = 0;
    for (int j = 0; j < 8; ++j) {
      float p = pv[j]; int idx = j * 16 + r;
      if (p > v1)      { v2 = v1; i2 = i1; v1 = p; i1 = idx; }
      else if (p > v2) { v2 = p; i2 = idx; }
    }
    for (int off = 1; off < 16; off <<= 1) {
      float ov1 = __shfl_xor(v1, off, 32), ov2 = __shfl_xor(v2, off, 32);
      int   oi1 = __shfl_xor(i1, off, 32), oi2 = __shfl_xor(i2, off, 32);
      if (ov1 > v1 || (ov1 == v1 && oi1 < i1)) { v2 = v1; i2 = i1; v1 = ov1; i1 = oi1; }
      else if (ov1 > v2 || (ov1 == v2 && oi1 < i2)) { v2 = ov1; i2 = oi1; }
      if (ov2 > v2 || (ov2 == v2 && oi2 < i2)) { v2 = ov2; i2 = oi2; }
    }
    float o = v1 * inv * vs[i1][r] + v2 * inv * vs[i2][r];
    xo[((size_t)b * kN + rg) * kC + head * kHD + r] = (__bf16)o;
  }
}

// ---------------------------------------------------------------------------
// Kernel 3: output projection  xo (bf16) @ Wproj + bproj -> f32 out
// ---------------------------------------------------------------------------
__global__ void proj_kernel(const __bf16* __restrict__ xo,
                            const __bf16* __restrict__ Wprojpack,
                            const float* __restrict__ bproj,
                            float* __restrict__ out) {
  __shared__ __bf16 xs[8][16 * 128];
  int tid = threadIdx.x, w = tid >> 5, lane = tid & 31;
  int rowbase = blockIdx.x * 128 + w * 16;

  const uint4* src = (const uint4*)(xo + (size_t)rowbase * kC);
  uint4* dst4 = (uint4*)xs[w];
  for (int i = lane; i < 16 * 128 * 2 / 16; i += 32) dst4[i] = src[i];
  __syncthreads();

  int sel = lane >> 4, r = lane & 15;
  v16bf A[4];
  for (int kc = 0; kc < 4; ++kc)
    for (int e = 0; e < 16; ++e)
      A[kc][e] = xs[w][r * 128 + kc * 32 + klocal(sel, e)];

  for (int t = 0; t < 8; ++t) {
    int tn = (t + 1 < 8) ? t + 1 : t;
    __builtin_prefetch(Wprojpack + (((size_t)tn * 4) * 32 + lane) * 16, 0, 0);
    v16bf Bv[4];
#pragma unroll
    for (int kc = 0; kc < 4; ++kc)
      Bv[kc] = *(const v16bf*)(Wprojpack + (((size_t)t * 4 + kc) * 32 + lane) * 16);
    v8f acc = {};
#pragma unroll
    for (int kc = 0; kc < 4; ++kc)
      acc = __builtin_amdgcn_wmma_f32_16x16x32_bf16(false, A[kc], false, Bv[kc],
                                                    (short)0, acc, false, false);
    int coln = t * 16 + r;
    float bp = bproj[coln];
#pragma unroll
    for (int i = 0; i < 8; ++i) {
      int rg = rowbase + i + sel * 8;
      out[(size_t)rg * kC + coln] = acc[i] + bp;
    }
  }
}

extern "C" void kernel_launch(void* const* d_in, const int* in_sizes, int n_in,
                              void* d_out, int out_size, void* d_ws, size_t ws_size,
                              hipStream_t stream) {
  (void)in_sizes; (void)n_in; (void)out_size; (void)ws_size;
  const float* x        = (const float*)d_in[0];
  const float* mask     = (const float*)d_in[1];
  const float* Wq       = (const float*)d_in[2];
  const float* Wkv      = (const float*)d_in[3];
  const float* Wproj    = (const float*)d_in[4];
  const float* bproj    = (const float*)d_in[5];
  const float* bias_tbl = (const float*)d_in[6];
  const int*   rel_idx  = (const int*)d_in[7];
  float* out = (float*)d_out;

  char* ws = (char*)d_ws;
  size_t off = 0;
  __bf16* qbf = (__bf16*)(ws + off); off += (size_t)kB * kN * kC * 2;
  __bf16* kbf = (__bf16*)(ws + off); off += (size_t)kB * kN * kC * 2;
  float*  vf  = (float*) (ws + off); off += (size_t)kB * kN * kC * 4;
  __bf16* xo  = (__bf16*)(ws + off); off += (size_t)kB * kN * kC * 2;
  float*  bias_full = (float*)(ws + off); off += (size_t)kH * kN * kN * 4;
  __bf16* Wpack     = (__bf16*)(ws + off); off += (size_t)24 * 4 * 32 * 16 * 2;
  __bf16* Wprojpack = (__bf16*)(ws + off); off += (size_t)8 * 4 * 32 * 16 * 2;

  int prep_threads = 24 * 4 * 32 + 8 * 4 * 32 + kH * kN * kN;
  prep_kernel<<<(prep_threads + 255) / 256, 256, 0, stream>>>(
      Wq, Wkv, Wproj, bias_tbl, rel_idx, Wpack, Wprojpack, bias_full);
  qkv_kernel<<<kB * kN / 128, 256, 0, stream>>>(x, Wpack, qbf, kbf, vf);
  attn_kernel<<<dim3(kH, kB), 256, 0, stream>>>(mask, bias_full, qbf, kbf, vf, xo);
  proj_kernel<<<kB * kN / 128, 256, 0, stream>>>(xo, Wprojpack, bproj, out);
}
